// LightCPRAformerC2f_54382875902245
// MI455X (gfx1250) — matrix operand order repaired and checked by reading, not compiled, not run on hardware
//
#include <hip/hip_runtime.h>
#include <math.h>

typedef __attribute__((ext_vector_type(16))) _Float16 v16h;
typedef __attribute__((ext_vector_type(8)))  float    v8f;

union Frag16 { v16h v; _Float16 e[16]; unsigned u[8]; };
union Frag8f { v8f  v; float     e[8];  };

#define HW   2304          // 48*48 tokens
#define EPSC 1e-5f

// pack two fp32 -> packed f16x2 in one u32: single v_cvt_pk_rz_f16_f32
__device__ __forceinline__ unsigned pkh2(float lo, float hi) {
    typedef __attribute__((ext_vector_type(2))) __fp16 fp16x2;
    union { fp16x2 h; unsigned u; } p;
    p.h = __builtin_amdgcn_cvt_pkrtz(lo, hi);
    return p.u;
}
// A-fragment (16-bit 16x32) u32 index for vgpr-pair j, given lane half.
__device__ __forceinline__ int a_u32_idx(int j, int half) {
    return half * 8 + j;
}

// ---------------------------------------------------------------------------
// Generic fused GEMM:  out[b, m, n] = epilogue( sum_k W[m,k] * X[b,k,n] )
// 256 threads (8 waves), block tile 128(M) x 64(N), K chunks of 32.
// Software pipelined: global->reg prefetch of chunk c+1 overlaps the 4 WMMAs
// of chunk c; ping-pong LDS buffers, ONE barrier per K chunk.
// modes: 0 BN+SiLU dual-write (cv1)   1 +bias (qkv)
//        2 +bias+residual (proj)      3 +bias, exact GELU (f1)
//        4 +bias+residual dual-write  5 BN+SiLU (cv2)
// ---------------------------------------------------------------------------
__global__ __launch_bounds__(256)
void gemm_wmma_epi(const float* __restrict__ X, long xBS,
                   const float* __restrict__ W, int K,
                   const float* __restrict__ bias,
                   const float* __restrict__ bn_g, const float* __restrict__ bn_b,
                   const float* __restrict__ bn_m, const float* __restrict__ bn_v,
                   const float* __restrict__ R, long rBS,
                   float* __restrict__ O, long oBS, int oCO,
                   float* __restrict__ O2, long o2BS, int o2CO,
                   int mode)
{
    __shared__ unsigned Wt[2][128][17];   // [buf][m][k2]
    __shared__ unsigned Xn[2][64][17];    // [buf][n][k2]

    const int tid  = threadIdx.x;
    const int lane = tid & 31;
    const int wv   = tid >> 5;
    const int mi   = wv >> 1;
    const int ni   = wv & 1;
    const int half = lane >> 4;
    const int l15  = lane & 15;
    const int m0   = blockIdx.y * 128;
    const int n0   = blockIdx.x * 64;
    const int b    = blockIdx.z;
    const float* Xb = X + (long)b * xBS;

    // staging coordinates
    const int r  = tid >> 1;              // W row 0..127
    const int cg = (tid & 1) * 8;         // W u32 offset 0/8
    const int k2 = tid >> 4;              // X k-pair 0..15
    const int ng = (tid & 15) * 4;        // X n offset
    const float* wrow = W  + (long)(m0 + r) * K + cg * 2;
    const float* xrow = Xb + (long)(2 * k2) * HW + n0 + ng;

    Frag8f acc[2][2];
#pragma unroll
    for (int mt = 0; mt < 2; ++mt)
#pragma unroll
        for (int nt = 0; nt < 2; ++nt)
#pragma unroll
            for (int j = 0; j < 8; ++j) acc[mt][nt].e[j] = 0.f;

    float wreg[16], xl[4], xh[4];
    // ---- prologue: load + stage chunk 0 ----
#pragma unroll
    for (int j = 0; j < 16; ++j) wreg[j] = wrow[j];
#pragma unroll
    for (int j = 0; j < 4; ++j) { xl[j] = xrow[j]; xh[j] = xrow[HW + j]; }
#pragma unroll
    for (int j = 0; j < 8; ++j) Wt[0][r][cg + j] = pkh2(wreg[2 * j], wreg[2 * j + 1]);
#pragma unroll
    for (int j = 0; j < 4; ++j) Xn[0][ng + j][k2] = pkh2(xl[j], xh[j]);
    __syncthreads();

    const int nchunks = K >> 5;
    for (int c = 0; c < nchunks; ++c) {
        const int cb = c & 1;
        // ---- prefetch chunk c+1 into registers (overlaps WMMAs) ----
        if (c + 1 < nchunks) {
            const float* wp = wrow + (c + 1) * 32;
            const float* xp = xrow + (long)(c + 1) * 32 * HW;
#pragma unroll
            for (int j = 0; j < 16; ++j) wreg[j] = wp[j];
#pragma unroll
            for (int j = 0; j < 4; ++j) { xl[j] = xp[j]; xh[j] = xp[HW + j]; }
        }
        // ---- fragments from LDS buffer cb: pure b32 DS reads ----
        Frag16 a0, a1, b0, b1;
#pragma unroll
        for (int j = 0; j < 8; ++j) {
            int ai = a_u32_idx(j, half);
            a0.u[j] = Wt[cb][mi * 32 + l15][ai];
            a1.u[j] = Wt[cb][mi * 32 + 16 + l15][ai];
            b0.u[j] = Xn[cb][ni * 32 + l15][half * 8 + j];
            b1.u[j] = Xn[cb][ni * 32 + 16 + l15][half * 8 + j];
        }
        acc[0][0].v = __builtin_amdgcn_wmma_f32_16x16x32_f16(false, a0.v, false, b0.v,
                                                             (short)0, acc[0][0].v, false, false);
        acc[0][1].v = __builtin_amdgcn_wmma_f32_16x16x32_f16(false, a0.v, false, b1.v,
                                                             (short)0, acc[0][1].v, false, false);
        acc[1][0].v = __builtin_amdgcn_wmma_f32_16x16x32_f16(false, a1.v, false, b0.v,
                                                             (short)0, acc[1][0].v, false, false);
        acc[1][1].v = __builtin_amdgcn_wmma_f32_16x16x32_f16(false, a1.v, false, b1.v,
                                                             (short)0, acc[1][1].v, false, false);
        // ---- stage chunk c+1 into the other buffer; single barrier ----
        if (c + 1 < nchunks) {
            const int nb = cb ^ 1;
#pragma unroll
            for (int j = 0; j < 8; ++j) Wt[nb][r][cg + j] = pkh2(wreg[2 * j], wreg[2 * j + 1]);
#pragma unroll
            for (int j = 0; j < 4; ++j) Xn[nb][ng + j][k2] = pkh2(xl[j], xh[j]);
        }
        __syncthreads();
    }

    // ---- epilogue: mode hoisted outside the unrolled store loops ----
    float*       Ob  = O  + (long)b * oBS;
    float*       O2b = O2 ? O2 + (long)b * o2BS : nullptr;
    const float* Rb  = R  ? R  + (long)b * rBS  : nullptr;
    const int mb = m0 + mi * 32;
    const int nb = n0 + ni * 32 + l15;

    if (mode == 0 || mode == 5) {
#pragma unroll
        for (int mt = 0; mt < 2; ++mt)
#pragma unroll
            for (int j = 0; j < 8; ++j) {
                int   om = mb + mt * 16 + j + 8 * half;
                float s  = bn_g[om] * rsqrtf(bn_v[om] + EPSC);
                float sh = bn_b[om] - bn_m[om] * s;
#pragma unroll
                for (int nt = 0; nt < 2; ++nt) {
                    float t = acc[mt][nt].e[j] * s + sh;
                    float y = t / (1.f + __expf(-t));                 // SiLU
                    Ob[(long)(oCO + om) * HW + nb + nt * 16] = y;
                    if (mode == 0 && om >= 128)
                        O2b[(long)(om - 128) * HW + nb + nt * 16] = y;
                }
            }
    } else if (mode == 1) {
#pragma unroll
        for (int mt = 0; mt < 2; ++mt)
#pragma unroll
            for (int j = 0; j < 8; ++j) {
                int   om = mb + mt * 16 + j + 8 * half;
                float bi = bias[om];
#pragma unroll
                for (int nt = 0; nt < 2; ++nt)
                    Ob[(long)om * HW + nb + nt * 16] = acc[mt][nt].e[j] + bi;
            }
    } else if (mode == 3) {
#pragma unroll
        for (int mt = 0; mt < 2; ++mt)
#pragma unroll
            for (int j = 0; j < 8; ++j) {
                int   om = mb + mt * 16 + j + 8 * half;
                float bi = bias[om];
#pragma unroll
                for (int nt = 0; nt < 2; ++nt) {
                    float t = acc[mt][nt].e[j] + bi;
                    Ob[(long)om * HW + nb + nt * 16] =
                        0.5f * t * (1.f + erff(t * 0.70710678118654752f));   // exact GELU
                }
            }
    } else { // mode 2 or 4: bias + residual (+ optional concat copy)
#pragma unroll
        for (int mt = 0; mt < 2; ++mt)
#pragma unroll
            for (int j = 0; j < 8; ++j) {
                int   om = mb + mt * 16 + j + 8 * half;
                float bi = bias[om];
#pragma unroll
                for (int nt = 0; nt < 2; ++nt) {
                    float y = acc[mt][nt].e[j] + bi + Rb[(long)om * HW + nb + nt * 16];
                    Ob[(long)om * HW + nb + nt * 16] = y;
                    if (mode == 4)
                        O2b[(long)(o2CO + om) * HW + nb + nt * 16] = y;
                }
            }
    }
}

// ---------------------------------------------------------------------------
// GroupNorm over all (C,H,W) per batch element, per-channel affine.
// ---------------------------------------------------------------------------
__global__ __launch_bounds__(256)
void groupnorm_kernel(const float* __restrict__ X, const float* __restrict__ g,
                      const float* __restrict__ bcoef, float* __restrict__ O)
{
    __shared__ float s1[256], s2[256];
    const int  b    = blockIdx.x;
    const long tot4 = 128l * HW / 4;                 // float4 count
    const float4* X4 = (const float4*)(X + (long)b * 128 * HW);
    float sum = 0.f, sq = 0.f;
    for (long i = threadIdx.x; i < tot4; i += 256) {
        float4 x = X4[i];
        sum += x.x + x.y + x.z + x.w;
        sq  += x.x * x.x + x.y * x.y + x.z * x.z + x.w * x.w;
    }
    s1[threadIdx.x] = sum; s2[threadIdx.x] = sq;
    __syncthreads();
    for (int st = 128; st > 0; st >>= 1) {
        if ((int)threadIdx.x < st) {
            s1[threadIdx.x] += s1[threadIdx.x + st];
            s2[threadIdx.x] += s2[threadIdx.x + st];
        }
        __syncthreads();
    }
    const float total = 128.f * HW;
    float mu  = s1[0] / total;
    float var = s2[0] / total - mu * mu;
    float inv = rsqrtf(var + EPSC);
    float4* O4 = (float4*)(O + (long)b * 128 * HW);
    for (long i = threadIdx.x; i < tot4; i += 256) {
        int c = (int)((i * 4) / HW);                 // HW % 4 == 0 -> uniform in vec
        float sc = inv * g[c], shf = bcoef[c] - mu * sc;
        float4 x = X4[i];
        float4 y;
        y.x = x.x * sc + shf; y.y = x.y * sc + shf;
        y.z = x.z * sc + shf; y.w = x.w * sc + shf;
        O4[i] = y;
    }
}

// ---------------------------------------------------------------------------
// Flash attention (reference quirk: q/k/v come from batches 0/1/2; qkv-split
// axis is the attention batch "g"). 4 waves/block share ping-ponged K/V
// staging; each wave owns one 16-row q-tile, online softmax over 32-key
// chunks, K/V global->reg prefetch overlapped with the 4 WMMAs per chunk.
// Output channel c = d*HEADS + h (reference transpose/reshape).
// ---------------------------------------------------------------------------
__global__ __launch_bounds__(128)
void flash_attn_kernel(const float* __restrict__ qkv, float* __restrict__ out)
{
    __shared__ unsigned Kt[2][32][17];    // [buf][key][d2]
    __shared__ unsigned Vt[2][32][17];    // [buf][d][key2]
    __shared__ unsigned Pt[4][16][17];    // per-wave P [row][key2]

    const int tid  = threadIdx.x;
    const int lane = tid & 31;
    const int wv   = tid >> 5;
    const int half = lane >> 4;
    const int l15  = lane & 15;
    const int gh   = blockIdx.y;
    const int g    = gh >> 2;
    const int h    = gh & 3;
    const int q0   = blockIdx.x * 64 + wv * 16;

    const long BS    = 384l * HW;
    const long chOff = (long)(g * 128 + h * 32) * HW;
    const float* Qp = qkv + 0 * BS + chOff;
    const float* Kp = qkv + 1 * BS + chOff;
    const float* Vp = qkv + 2 * BS + chOff;

    // staging coordinates
    const int kkey = tid & 31;            // K: key lane, d2 strided by 4
    const int kd2b = tid >> 5;
    const int vd   = tid >> 2;            // V: d row
    const int vkq  = (tid & 3) * 4;       // V: key2 offset

    // Q as A-fragment (16 rows x 32 dims), once per wave.
    Frag16 qa;
#pragma unroll
    for (int j = 0; j < 8; ++j) {
        int d = half * 16 + 2 * j;
        qa.u[j] = pkh2(Qp[(long)d * HW + q0 + l15], Qp[(long)(d + 1) * HW + q0 + l15]);
    }

    Frag8f o0, o1, zf;
    float rm[8], rl[8];
#pragma unroll
    for (int j = 0; j < 8; ++j) {
        o0.e[j] = 0.f; o1.e[j] = 0.f; zf.e[j] = 0.f;
        rm[j] = -3.0e38f; rl[j] = 0.f;
    }
    const float scale = 0.17677669529663687f;      // 32^-0.5

    float kr[4][2], vr[4][2];
    // ---- prologue: load + stage chunk 0 ----
#pragma unroll
    for (int i = 0; i < 4; ++i) {
        int d2 = kd2b + 4 * i;
        kr[i][0] = Kp[(long)(2 * d2) * HW + kkey];
        kr[i][1] = Kp[(long)(2 * d2 + 1) * HW + kkey];
        vr[i][0] = Vp[(long)vd * HW + 2 * (vkq + i)];
        vr[i][1] = Vp[(long)vd * HW + 2 * (vkq + i) + 1];
    }
#pragma unroll
    for (int i = 0; i < 4; ++i) {
        Kt[0][kkey][kd2b + 4 * i] = pkh2(kr[i][0], kr[i][1]);
        Vt[0][vd][vkq + i]        = pkh2(vr[i][0], vr[i][1]);
    }
    __syncthreads();

    const int nchunks = HW / 32;          // 72
    for (int c = 0; c < nchunks; ++c) {
        const int cb = c & 1;
        // ---- prefetch next K/V chunk into registers ----
        if (c + 1 < nchunks) {
            const int key0 = (c + 1) * 32;
#pragma unroll
            for (int i = 0; i < 4; ++i) {
                int d2 = kd2b + 4 * i;
                kr[i][0] = Kp[(long)(2 * d2) * HW + key0 + kkey];
                kr[i][1] = Kp[(long)(2 * d2 + 1) * HW + key0 + kkey];
                vr[i][0] = Vp[(long)vd * HW + key0 + 2 * (vkq + i)];
                vr[i][1] = Vp[(long)vd * HW + key0 + 2 * (vkq + i) + 1];
            }
        }

        // ---- S = Q.K^T ----
        Frag16 b0, b1;
#pragma unroll
        for (int j = 0; j < 8; ++j) {
            b0.u[j] = Kt[cb][l15][half * 8 + j];
            b1.u[j] = Kt[cb][16 + l15][half * 8 + j];
        }
        Frag8f s0, s1;
        s0.v = __builtin_amdgcn_wmma_f32_16x16x32_f16(false, qa.v, false, b0.v,
                                                      (short)0, zf.v, false, false);
        s1.v = __builtin_amdgcn_wmma_f32_16x16x32_f16(false, qa.v, false, b1.v,
                                                      (short)0, zf.v, false, false);

        // ---- online softmax; row m = j + 8*half within one 16-lane half ----
        _Float16* prow_base = (_Float16*)&Pt[wv][0][0];
#pragma unroll
        for (int j = 0; j < 8; ++j) {
            float a0 = s0.e[j] * scale;
            float a1 = s1.e[j] * scale;
            float t  = fmaxf(a0, a1);
            t = fmaxf(t, __shfl_xor(t, 1, 32));
            t = fmaxf(t, __shfl_xor(t, 2, 32));
            t = fmaxf(t, __shfl_xor(t, 4, 32));
            t = fmaxf(t, __shfl_xor(t, 8, 32));
            float nm    = fmaxf(rm[j], t);
            float alpha = __expf(rm[j] - nm);
            rm[j] = nm;
            float p0 = __expf(a0 - nm);
            float p1 = __expf(a1 - nm);
            float rs = p0 + p1;
            rs += __shfl_xor(rs, 1, 32);
            rs += __shfl_xor(rs, 2, 32);
            rs += __shfl_xor(rs, 4, 32);
            rs += __shfl_xor(rs, 8, 32);
            rl[j] = rl[j] * alpha + rs;
            o0.e[j] *= alpha;
            o1.e[j] *= alpha;
            _Float16* prow = prow_base + (j + 8 * half) * 34;   // 17 u32 = 34 halves
            prow[l15]      = (_Float16)p0;
            prow[16 + l15] = (_Float16)p1;
        }
        // P is wave-private: DS-counter ordering is enough (no block barrier)
        asm volatile("s_wait_dscnt 0x0" ::: "memory");

        // ---- O += P.V ----
        Frag16 ap, bv0, bv1;
#pragma unroll
        for (int j = 0; j < 8; ++j) {
            ap.u[j]  = Pt[wv][l15][a_u32_idx(j, half)];
            bv0.u[j] = Vt[cb][l15][half * 8 + j];
            bv1.u[j] = Vt[cb][16 + l15][half * 8 + j];
        }
        o0.v = __builtin_amdgcn_wmma_f32_16x16x32_f16(false, ap.v, false, bv0.v,
                                                      (short)0, o0.v, false, false);
        o1.v = __builtin_amdgcn_wmma_f32_16x16x32_f16(false, ap.v, false, bv1.v,
                                                      (short)0, o1.v, false, false);

        // ---- stage next chunk into the other buffer; single barrier ----
        if (c + 1 < nchunks) {
            const int nbuf = cb ^ 1;
#pragma unroll
            for (int i = 0; i < 4; ++i) {
                Kt[nbuf][kkey][kd2b + 4 * i] = pkh2(kr[i][0], kr[i][1]);
                Vt[nbuf][vd][vkq + i]        = pkh2(vr[i][0], vr[i][1]);
            }
        }
        __syncthreads();
    }

    // write out[g, d*4 + h, q]
    float* Ob = out + (long)g * 128 * HW;
#pragma unroll
    for (int j = 0; j < 8; ++j) {
        int   q   = q0 + j + 8 * half;
        float inv = 1.f / rl[j];
        Ob[(long)(l15 * 4 + h) * HW + q]        = o0.e[j] * inv;
        Ob[(long)((16 + l15) * 4 + h) * HW + q] = o1.e[j] * inv;
    }
}

// ---------------------------------------------------------------------------
extern "C" void kernel_launch(void* const* d_in, const int* in_sizes, int n_in,
                              void* d_out, int out_size, void* d_ws, size_t ws_size,
                              hipStream_t stream)
{
    const float* x      = (const float*)d_in[0];
    const float* cv1_w  = (const float*)d_in[1];
    const float* cv1_g  = (const float*)d_in[2];
    const float* cv1_b  = (const float*)d_in[3];
    const float* cv1_m  = (const float*)d_in[4];
    const float* cv1_v  = (const float*)d_in[5];
    const float* n1_g   = (const float*)d_in[6];
    const float* n1_b   = (const float*)d_in[7];
    const float* qkv_w  = (const float*)d_in[8];
    const float* qkv_b  = (const float*)d_in[9];
    const float* proj_w = (const float*)d_in[10];
    const float* proj_b = (const float*)d_in[11];
    const float* n2_g   = (const float*)d_in[12];
    const float* n2_b   = (const float*)d_in[13];
    const float* f1_w   = (const float*)d_in[14];
    const float* f1_b   = (const float*)d_in[15];
    const float* f2_w   = (const float*)d_in[16];
    const float* f2_b   = (const float*)d_in[17];
    const float* cv2_w  = (const float*)d_in[18];
    const float* cv2_g  = (const float*)d_in[19];
    const float* cv2_bb = (const float*)d_in[20];
    const float* cv2_m  = (const float*)d_in[21];
    const float* cv2_v  = (const float*)d_in[22];

    const long HWl = 2304;
    float* ws   = (float*)d_ws;
    float* cat  = ws;                      // (3, 512, HW)  concat target
    float* cur  = cat  + 3 * 512 * HWl;    // (3, 128, HW)
    float* tmpN = cur  + 3 * 128 * HWl;    // (3, 128, HW)
    float* qkvb = tmpN + 3 * 128 * HWl;    // (3, 384, HW)
    float* attn = qkvb + 3 * 384 * HWl;    // (3, 128, HW)
    float* ffb  = attn + 3 * 128 * HWl;    // (3, 256, HW)

    dim3 blk(256);

    // cv1: (256x256) GEMM + BN + SiLU -> cat[:,0:256]; also cur = y2 copy
    gemm_wmma_epi<<<dim3(36, 2, 3), blk, 0, stream>>>(
        x, 256 * HWl, cv1_w, 256,
        nullptr, cv1_g, cv1_b, cv1_m, cv1_v, nullptr, 0,
        cat, 512 * HWl, 0, cur, 128 * HWl, 0, 0);

    for (int i = 0; i < 2; ++i) {
        groupnorm_kernel<<<dim3(3), dim3(256), 0, stream>>>(
            cur, n1_g + i * 128, n1_b + i * 128, tmpN);
        // qkv: (384x128) GEMM + bias
        gemm_wmma_epi<<<dim3(36, 3, 3), blk, 0, stream>>>(
            tmpN, 128 * HWl, qkv_w + (long)i * 384 * 128, 128,
            qkv_b + i * 384, nullptr, nullptr, nullptr, nullptr, nullptr, 0,
            qkvb, 384 * HWl, 0, nullptr, 0, 0, 1);
        flash_attn_kernel<<<dim3(36, 12), dim3(128), 0, stream>>>(qkvb, attn);
        // proj: (128x128) GEMM + bias + residual -> cur (in place)
        gemm_wmma_epi<<<dim3(36, 1, 3), blk, 0, stream>>>(
            attn, 128 * HWl, proj_w + (long)i * 128 * 128, 128,
            proj_b + i * 128, nullptr, nullptr, nullptr, nullptr, cur, 128 * HWl,
            cur, 128 * HWl, 0, nullptr, 0, 0, 2);
        groupnorm_kernel<<<dim3(3), dim3(256), 0, stream>>>(
            cur, n2_g + i * 128, n2_b + i * 128, tmpN);
        // f1: (256x128) GEMM + bias + exact GELU
        gemm_wmma_epi<<<dim3(36, 2, 3), blk, 0, stream>>>(
            tmpN, 128 * HWl, f1_w + (long)i * 256 * 128, 128,
            f1_b + i * 256, nullptr, nullptr, nullptr, nullptr, nullptr, 0,
            ffb, 256 * HWl, 0, nullptr, 0, 0, 3);
        // f2: (128x256) GEMM + bias + residual -> cur and cat slice
        gemm_wmma_epi<<<dim3(36, 1, 3), blk, 0, stream>>>(
            ffb, 256 * HWl, f2_w + (long)i * 128 * 256, 256,
            f2_b + i * 128, nullptr, nullptr, nullptr, nullptr, cur, 128 * HWl,
            cur, 128 * HWl, 0, cat, 512 * HWl, 256 + 128 * i, 4);
    }

    // cv2: (256x512) GEMM + BN + SiLU -> d_out
    gemm_wmma_epi<<<dim3(36, 2, 3), blk, 0, stream>>>(
        cat, 512 * HWl, cv2_w, 512,
        nullptr, cv2_g, cv2_bb, cv2_m, cv2_v, nullptr, 0,
        (float*)d_out, 256 * HWl, 0, nullptr, 0, 0, 5);
}
